// HyperedgeAttention_3959959847391
// MI455X (gfx1250) — matrix operand order repaired and checked by reading, not compiled, not run on hardware
//
#include <hip/hip_runtime.h>

// ---------------- problem constants ----------------
#define N_ROWS 20000
#define NPAD   20032            // N padded to 32 (zero-filled tail in gT)
#define N_E    5000
#define DIMD   128
#define EGRPS  157              // ceil(5000/32)
#define EPAD   (EGRPS * 32)     // 5024
#define NSLICE 8
#define SLICE_ROWS (N_ROWS / NSLICE)   // 2500 = 78*32 + 4

typedef __attribute__((ext_vector_type(8)))  float    v8f;
typedef __attribute__((ext_vector_type(4)))  float    f4;
typedef __attribute__((ext_vector_type(4)))  unsigned u4;
typedef __attribute__((ext_vector_type(16))) __bf16   v16bf;
typedef __attribute__((ext_vector_type(8)))  __bf16   v8bf;

// ---------------- workspace layout (bytes) ----------------
constexpr size_t OFF_H    = 0;              // h       : 20000*128*4 = 10,240,000
constexpr size_t OFF_SC   = 10240000;       // sc      : 80,000
constexpr size_t OFF_M    = 10320000;       // max     : 256 (padded)
constexpr size_t OFF_P    = 10320256;       // p       : 80,000
constexpr size_t OFF_GT   = 10400256;       // gT bf16 : 128*20032*2 = 5,128,192
constexpr size_t OFF_BITS = 15528448;       // bits    : 157*20000*4 = 12,560,000
constexpr size_t OFF_ZP   = 28088448;       // Zpart   : 8*5024*4 = 160,768
constexpr size_t OFF_OP   = 28249216;       // outpart : 8*5024*128*4 = 20,578,304
constexpr size_t OFF_RZ   = 48827520;       // rZ      : 5024*4
                                            // total  ~= 48.85 MB

__device__ inline unsigned ballot32(int pred) {
  return __builtin_amdgcn_ballot_w32(pred != 0);
}

// ============ K0: h = x @ W.T  and  sc = (h@u)/sqrt(128) ============
__global__ __launch_bounds__(256) void k_h_sc(const float* __restrict__ x,
                                              const float* __restrict__ W,
                                              const float* __restrict__ u,
                                              float* __restrict__ hG,
                                              float* __restrict__ scG) {
  __shared__ float xs[16 * 128];
  __shared__ float red[16 * 128];
  const int t  = threadIdx.x;
  const int n0 = blockIdx.x * 16;
  #pragma unroll
  for (int i = 0; i < 8; ++i) xs[t + i * 256] = x[(size_t)n0 * 128 + t + i * 256];
  __syncthreads();
  const int d  = t & 127;
  const int rh = t >> 7;
  float acc[8] = {0.f, 0.f, 0.f, 0.f, 0.f, 0.f, 0.f, 0.f};
  for (int k = 0; k < 128; ++k) {
    float wv = W[d * 128 + k];
    #pragma unroll
    for (int r = 0; r < 8; ++r) acc[r] += xs[(rh * 8 + r) * 128 + k] * wv;
  }
  const float uv = u[d];
  #pragma unroll
  for (int r = 0; r < 8; ++r) {
    int row = rh * 8 + r;
    hG[(size_t)(n0 + row) * 128 + d] = acc[r];
    red[row * 128 + d] = acc[r] * uv;
  }
  __syncthreads();
  if (t < 16) {
    float s = 0.f;
    for (int dd = 0; dd < 128; ++dd) s += red[t * 128 + dd];
    scG[n0 + t] = s * 0.08838834764831845f;  // 1/sqrt(128)
  }
}

// ============ K1: global max of sc (single block -> deterministic) ============
__global__ __launch_bounds__(1024) void k_max(const float* __restrict__ sc,
                                              float* __restrict__ Mout) {
  __shared__ float red[1024];
  float m = -3.4e38f;
  for (int i = threadIdx.x; i < N_ROWS; i += 1024) m = fmaxf(m, sc[i]);
  red[threadIdx.x] = m;
  __syncthreads();
  for (int s = 512; s > 0; s >>= 1) {
    if (threadIdx.x < s) red[threadIdx.x] = fmaxf(red[threadIdx.x], red[threadIdx.x + s]);
    __syncthreads();
  }
  if (threadIdx.x == 0) *Mout = red[0];
}

// ============ K2: p[n] = exp(sc[n] - M) ============
__global__ __launch_bounds__(256) void k_p(const float* __restrict__ sc,
                                           const float* __restrict__ M,
                                           float* __restrict__ p) {
  int i = blockIdx.x * 256 + threadIdx.x;
  if (i < N_ROWS) p[i] = __expf(sc[i] - *M);
}

// ============ K2b: gT[d][n] = bf16(p[n] * h[n][d]), zero-padded to NPAD ======
__global__ __launch_bounds__(256) void k_gT(const float* __restrict__ hG,
                                            const float* __restrict__ p,
                                            __bf16* __restrict__ gT) {
  __shared__ __align__(16) __bf16 tile[128 * 32];   // [d][k]
  const int t  = threadIdx.x;
  const int n0 = blockIdx.x * 32;
  {
    int k  = t >> 3;               // 0..31
    int dg = (t & 7) * 16;         // d group of 16
    int n  = n0 + k;
    if (n < N_ROWS) {
      float pv = p[n];
      const f4* hp = (const f4*)(hG + (size_t)n * 128 + dg);
      #pragma unroll
      for (int q = 0; q < 4; ++q) {
        f4 hv = hp[q];
        #pragma unroll
        for (int c = 0; c < 4; ++c)
          tile[(dg + q * 4 + c) * 32 + k] = (__bf16)(pv * hv[c]);
      }
    } else {
      #pragma unroll
      for (int i = 0; i < 16; ++i) tile[(dg + i) * 32 + k] = (__bf16)0.0f;
    }
  }
  __syncthreads();
  {
    int d = t >> 1, q = t & 1;
    const v8bf* src = (const v8bf*)&tile[d * 32 + q * 16];
    v8bf* dst = (v8bf*)(gT + (size_t)d * NPAD + n0 + q * 16);
    dst[0] = src[0];
    dst[1] = src[1];
  }
}

// ---- async copy of one 128d x 32k bf16 chunk of gT into LDS (8 KB) ----
__device__ inline void issue_chunk_async(const __bf16* __restrict__ gT,
                                         unsigned ldsbase, int n0, int t) {
  int d  = t >> 1;
  int q0 = (t & 1) * 2;                       // 0 or 2 (16B units)
  unsigned long long ga =
      (unsigned long long)(const void*)(gT + (size_t)d * NPAD + n0 + q0 * 8);
  unsigned la = ldsbase + (unsigned)(d * 64 + q0 * 16);
  asm volatile("global_load_async_to_lds_b128 %0, %1, off"
               :: "v"(la), "v"(ga) : "memory");
  asm volatile("global_load_async_to_lds_b128 %0, %1, off offset:16"
               :: "v"(la), "v"(ga) : "memory");
}

// ============ K3: main pass — ballot-pack mask, Z partials, WMMA partials ====
// grid (157 e-groups, 8 N-slices), block 256 (8 waves)
__global__ __launch_bounds__(256) void k_pass1(const int* __restrict__ H,
                                               const __bf16* __restrict__ gT,
                                               const float* __restrict__ p,
                                               unsigned* __restrict__ bits,
                                               float* __restrict__ Zpart,
                                               float* __restrict__ outpart) {
  __shared__ __align__(16) __bf16 gbuf[2][128 * 32];  // double-buffered [d][k]
  __shared__ __align__(16) unsigned mbits[32];
  __shared__ float zred[8][32];

  const int t    = threadIdx.x;
  const int w    = t >> 5;
  const int lane = t & 31;
  const int egrp = blockIdx.x;
  const int slc  = blockIdx.y;
  const int e0   = egrp * 32;
  const int nb   = slc * SLICE_ROWS;
  const int nfull = SLICE_ROWS / 32;         // 78 guard-free chunks
  const int nc    = nfull + 1;               // +1 tail chunk (4 rows)

  const int ml   = lane & 15;
  const int hi   = lane >> 4;
  const int esub = w & 1;
  const int d0   = (w >> 1) * 32;

  const int e   = e0 + lane;
  const int ecl = (e < N_E) ? e : (N_E - 1); // clamp, validity folded into pred
  const int ev  = (e < N_E);

  const unsigned lds0 = (unsigned)(size_t)&gbuf[0][0];
  const unsigned lds1 = (unsigned)(size_t)&gbuf[1][0];

  float zacc = 0.f;
  v8f acc0 = {};
  v8f acc1 = {};

  issue_chunk_async(gT, lds0, nb, t);        // prologue: chunk 0 -> buf 0

  for (int c = 0; c < nc; ++c) {
    const int cur = c & 1;
    const int n0  = nb + c * 32;
    const bool more = (c + 1) < nc;
    if (more) issue_chunk_async(gT, cur ? lds0 : lds1, n0 + 32, t);

    if (c < nfull) {
      // ---- full chunk: batch 4 H-row loads + 4 p loads, then ballots ----
      const int nr = n0 + w * 4;
      int   hv[4];
      float pv[4];
      #pragma unroll
      for (int r = 0; r < 4; ++r) hv[r] = H[(size_t)(nr + r) * N_E + ecl];
      #pragma unroll
      for (int r = 0; r < 4; ++r) pv[r] = p[nr + r];
      u4 wv;
      #pragma unroll
      for (int r = 0; r < 4; ++r) {
        int prd = (hv[r] > 0) & ev;
        wv[r] = ballot32(prd);
        zacc += prd ? pv[r] : 0.f;
      }
      if (lane == 0) {
        *(u4*)&mbits[w * 4] = wv;                                    // ds b128
        *(u4*)(bits + (size_t)egrp * N_ROWS + nr) = wv;              // global b128
      }
      if (c + 1 < nfull) {                   // wave-uniform prefetch guard
        #pragma unroll
        for (int r = 0; r < 4; ++r)
          __builtin_prefetch(H + (size_t)(nr + 32 + r) * N_E + ecl, 0, 0);
      }
    } else {
      // ---- tail chunk: 4 valid rows, wave 0 only; rows 4..31 zeroed ----
      if (w == 0) {
        mbits[lane] = 0u;                    // zero all 32 rows (same-wave order)
        int   hv[4];
        float pv[4];
        #pragma unroll
        for (int r = 0; r < 4; ++r) hv[r] = H[(size_t)(n0 + r) * N_E + ecl];
        #pragma unroll
        for (int r = 0; r < 4; ++r) pv[r] = p[n0 + r];
        u4 wv;
        #pragma unroll
        for (int r = 0; r < 4; ++r) {
          int prd = (hv[r] > 0) & ev;
          wv[r] = ballot32(prd);
          zacc += prd ? pv[r] : 0.f;
        }
        if (lane == 0) {
          *(u4*)&mbits[0] = wv;
          *(u4*)(bits + (size_t)egrp * N_ROWS + n0) = wv;
        }
      }
    }

    // ---- wait for current chunk's async copy (loads complete in order) ----
    if (more) asm volatile("s_wait_asynccnt 0x2" ::: "memory");
    else      asm volatile("s_wait_asynccnt 0x0" ::: "memory");
    __syncthreads();

    // ---- A fragment: mask bits -> exact bf16 0/1 (contiguous k = 16*half + j, same order as B) ----
    v16bf a;
    {
      int ebit = esub * 16 + ml;
      #pragma unroll
      for (int j = 0; j < 16; ++j) {
        int k = hi * 16 + j;
        unsigned bit = (mbits[k] >> ebit) & 1u;
        a[j] = (__bf16)(float)bit;
      }
    }
    // ---- B fragments (32x16 bf16, K split by lane half) + WMMA ----
    {
      const __bf16* bp = &gbuf[cur][(d0 + ml) * 32 + hi * 16];
      v8bf blo = *(const v8bf*)(bp);
      v8bf bhi = *(const v8bf*)(bp + 8);
      v16bf b;
      #pragma unroll
      for (int j = 0; j < 8; ++j) { b[j] = blo[j]; b[j + 8] = bhi[j]; }
      acc0 = __builtin_amdgcn_wmma_f32_16x16x32_bf16(false, a, false, b,
                                                     (short)0, acc0, false, false);
    }
    {
      const __bf16* bp = &gbuf[cur][(d0 + 16 + ml) * 32 + hi * 16];
      v8bf blo = *(const v8bf*)(bp);
      v8bf bhi = *(const v8bf*)(bp + 8);
      v16bf b;
      #pragma unroll
      for (int j = 0; j < 8; ++j) { b[j] = blo[j]; b[j + 8] = bhi[j]; }
      acc1 = __builtin_amdgcn_wmma_f32_16x16x32_bf16(false, a, false, b,
                                                     (short)0, acc1, false, false);
    }
    __syncthreads();   // everyone done reading gbuf[cur] & mbits
  }

  // ---- cross-wave Z reduction ----
  zred[w][lane] = zacc;
  __syncthreads();
  if (w == 0) {
    float z = 0.f;
    #pragma unroll
    for (int s = 0; s < 8; ++s) z += zred[s][lane];
    Zpart[(size_t)slc * EPAD + e0 + lane] = z;
  }

  // ---- store out partials (C/D layout: VGPR r -> M=r / M=r+8 by lane half) ----
  #pragma unroll
  for (int r = 0; r < 8; ++r) {
    int ee = e0 + esub * 16 + r + hi * 8;
    int dd = d0 + ml;
    outpart[((size_t)slc * EPAD + ee) * 128 + dd]      = acc0[r];
    outpart[((size_t)slc * EPAD + ee) * 128 + dd + 16] = acc1[r];
  }
}

// ============ K4: reduce slice partials -> out, and rZ = 1/Z ============
__global__ __launch_bounds__(256) void k_finalize(const float* __restrict__ Zpart,
                                                  const float* __restrict__ outpart,
                                                  float* __restrict__ outG,
                                                  float* __restrict__ rZ) {
  int idx = blockIdx.x * 256 + threadIdx.x;   // < 5000*128
  int e = idx >> 7, d = idx & 127;
  float Z = 0.f;
  #pragma unroll
  for (int s = 0; s < 8; ++s) Z += Zpart[(size_t)s * EPAD + e];
  float rz = (Z > 0.f) ? (1.f / Z) : 0.f;
  float o = 0.f;
  #pragma unroll
  for (int s = 0; s < 8; ++s) o += outpart[((size_t)s * EPAD + e) * 128 + d];
  outG[idx] = o * rz;
  if (d == 0) rZ[e] = rz;
}

// ============ K5: stream A with b128 stores (E divisible by 4) ============
__global__ __launch_bounds__(256) void k_writeA(const unsigned* __restrict__ bits,
                                                const float* __restrict__ p,
                                                const float* __restrict__ rZ,
                                                float* __restrict__ A) {
  size_t idx4 = ((size_t)blockIdx.x * 256 + threadIdx.x) * 4;
  if (idx4 >= (size_t)N_ROWS * N_E) return;
  int n = (int)(idx4 / N_E);
  int e = (int)(idx4 - (size_t)n * N_E);      // multiple of 4; e..e+3 same row/word
  unsigned word = bits[(size_t)(e >> 5) * N_ROWS + n];
  float pn = p[n];
  int b = e & 31;
  f4 v;
  #pragma unroll
  for (int c = 0; c < 4; ++c)
    v[c] = ((word >> (b + c)) & 1u) ? pn * rZ[e + c] : 0.f;
  *(f4*)(A + idx4) = v;
}

// ---------------- host launcher ----------------
extern "C" void kernel_launch(void* const* d_in, const int* in_sizes, int n_in,
                              void* d_out, int out_size, void* d_ws, size_t ws_size,
                              hipStream_t stream) {
  (void)in_sizes; (void)n_in; (void)out_size; (void)ws_size;
  const float* x = (const float*)d_in[0];
  const int*   H = (const int*)d_in[1];
  const float* W = (const float*)d_in[2];
  const float* u = (const float*)d_in[3];

  char* ws = (char*)d_ws;
  float*    hG      = (float*)(ws + OFF_H);
  float*    scG     = (float*)(ws + OFF_SC);
  float*    Mg      = (float*)(ws + OFF_M);
  float*    pG      = (float*)(ws + OFF_P);
  __bf16*   gT      = (__bf16*)(ws + OFF_GT);
  unsigned* bits    = (unsigned*)(ws + OFF_BITS);
  float*    Zpart   = (float*)(ws + OFF_ZP);
  float*    outpart = (float*)(ws + OFF_OP);
  float*    rZ      = (float*)(ws + OFF_RZ);

  float* outG = (float*)d_out;                 // [E,128]
  float* Aout = outG + (size_t)N_E * DIMD;     // [N,E]

  k_h_sc<<<dim3(N_ROWS / 16), dim3(256), 0, stream>>>(x, W, u, hG, scG);
  k_max<<<dim3(1), dim3(1024), 0, stream>>>(scG, Mg);
  k_p<<<dim3((N_ROWS + 255) / 256), dim3(256), 0, stream>>>(scG, Mg, pG);
  k_gT<<<dim3(NPAD / 32), dim3(256), 0, stream>>>(hG, pG, gT);
  k_pass1<<<dim3(EGRPS, NSLICE), dim3(256), 0, stream>>>(H, gT, pG, bits, Zpart, outpart);
  k_finalize<<<dim3((N_E * DIMD) / 256), dim3(256), 0, stream>>>(Zpart, outpart, outG, rZ);
  k_writeA<<<dim3((N_ROWS * (N_E / 4) + 255) / 256), dim3(256), 0, stream>>>(bits, pG, rZ, Aout);
}
